// RecurrentGCN_56762287784218
// MI455X (gfx1250) — compile-verified, matrix operand order checked by reading, not run on hardware
//
#include <hip/hip_runtime.h>
#include <hip/hip_bf16.h>

typedef __attribute__((ext_vector_type(16))) __bf16 v16bf;
typedef __attribute__((ext_vector_type(4)))  __bf16 v4bf;
typedef __attribute__((ext_vector_type(8)))  float  v8f;
typedef __attribute__((ext_vector_type(4)))  unsigned int u32x4;
typedef __attribute__((ext_vector_type(8)))  int i32x8;
typedef __attribute__((ext_vector_type(4)))  int i32x4;

#define D_IN 128
#define XS_LD 130   // f32 LDS row stride (128 data + 2 pad DWORDs) -> bank-conflict-free column reads

#if __has_builtin(__builtin_amdgcn_tensor_load_to_lds) && __has_builtin(__builtin_amdgcn_s_wait_tensorcnt)
#define USE_TDM 1
#else
#define USE_TDM 0
#endif

// ---------------------------------------------------------------- zero fill
__global__ void zero_kernel(float* p, size_t n) {
    size_t i = (size_t)blockIdx.x * blockDim.x + threadIdx.x;
    size_t stride = (size_t)gridDim.x * blockDim.x;
    for (; i < n; i += stride) p[i] = 0.0f;
}

// ---------------------------------------------------------------- degree count
__global__ void deg_kernel(const int* __restrict__ dst, float* __restrict__ deg, int E) {
    int i = blockIdx.x * blockDim.x + threadIdx.x;
    if (i < E) unsafeAtomicAdd(&deg[dst[i]], 1.0f);
}

// ---------------------------------------------------------------- dinv = (deg+2)^-1/2
__global__ void dinv_kernel(const float* __restrict__ deg, float* __restrict__ dinv, int N) {
    int i = blockIdx.x * blockDim.x + threadIdx.x;
    if (i < N) dinv[i] = rsqrtf(deg[i] + 2.0f);
}

// ---------------------------------------------------------------- M = W @ lW[:128,:]  (128x128, bf16 out)
__global__ void fuse_w_kernel(const float* __restrict__ W, const float* __restrict__ lW,
                              __bf16* __restrict__ out) {
    int idx = blockIdx.x * blockDim.x + threadIdx.x;   // 0..16383
    int i = idx >> 7, j = idx & 127;
    float acc = 0.0f;
#pragma unroll 8
    for (int m = 0; m < 128; ++m) acc += W[i * 128 + m] * lW[m * 128 + j];
    out[idx] = (__bf16)acc;
}

#if USE_TDM
// Issue one TDM load of a 16x128 f32 tile of x into LDS (padded rows: stride 130 DW).
// D# bit layout per CDNA5 ISA ch.8 (group0: count/lds_addr/global_addr/type,
// group1: data_size, pad, tensor/tile dims & strides). 2-D tensor, OOB rows read 0.
// 6-arg builtin form (clang-23 / therock-10.0 lane).
__device__ __forceinline__ void tdm_load_tile(const float* gsrc, unsigned lds_byte_addr,
                                              int rows_left) {
    unsigned long long ga = (unsigned long long)(uintptr_t)gsrc;
    u32x4 g0;
    g0[0] = 1u;                                            // count=1, user mode
    g0[1] = lds_byte_addr;                                 // lds_addr [63:32]
    g0[2] = (unsigned)(ga & 0xffffffffu);                  // global_addr lo
    g0[3] = (unsigned)((ga >> 32) & 0x01ffffffu) | 0x80000000u; // global_addr hi | type=2
    i32x8 g1;
    g1[0] = (int)(0x00020000u      // data_size = 2 (4 bytes)
                | (1u << 20)       // pad_enable
                | (6u << 22)       // pad_interval: 128 DWORDs
                | (1u << 25));     // pad_amount: 2 DWORDs
    g1[1] = (int)((128u & 0xffffu) << 16);                 // tensor_dim0[15:0] @ bit48
    g1[2] = (int)(((unsigned)rows_left & 0xffffu) << 16);  // tensor_dim1[15:0] @ bit80
    g1[3] = (int)((128u << 16) | (((unsigned)rows_left >> 16) & 0xffffu)); // tile_dim0 | tensor_dim1 hi
    g1[4] = 16;                                            // tile_dim1 = 16 rows
    g1[5] = 128;                                           // tensor_dim0_stride lo
    g1[6] = 0;
    g1[7] = 0;
    i32x4 gz4 = {0, 0, 0, 0};
    i32x8 gz8 = {0, 0, 0, 0, 0, 0, 0, 0};
    __builtin_amdgcn_tensor_load_to_lds(g0, g1, gz4, gz4, gz8, 0);
}
#endif

// ---------------------------------------------------------------- gz = x@Mz, gh = x@Mh (WMMA bf16)
// block = 256 thr = 8 waves; waves 0..3 -> Mz n-tile pairs, waves 4..7 -> Mh.
// Each block iterates 4 row-tiles of 16 rows; x tiles double-buffered in LDS via TDM.
__global__ void __launch_bounds__(256)
gemm_kernel(const float* __restrict__ x,
            const __bf16* __restrict__ Mz, const __bf16* __restrict__ Mh,
            __bf16* __restrict__ gz, __bf16* __restrict__ gh,
            int Nn, int nRowTiles) {
    __shared__ float xs[2][16 * XS_LD];

    const int tid  = threadIdx.x;
    const int wave = tid >> 5;
    const int lane = tid & 31;
    const int half = lane >> 4;      // 0: lanes 0-15, 1: lanes 16-31
    const int nrow = lane & 15;      // row (A) / column (B) within tile

    const __bf16* M = (wave < 4) ? Mz : Mh;
    __bf16*       G = (wave < 4) ? gz : gh;
    const int nt0 = (wave & 3) * 2;  // this wave's first n-tile (of 8)

    // ---- B fragments: resident in registers for the whole block
    v16bf bfrag[2][4];
#pragma unroll
    for (int t = 0; t < 2; ++t) {
        const int ncol = (nt0 + t) * 16 + nrow;
#pragma unroll
        for (int kt = 0; kt < 4; ++kt) {
#pragma unroll
            for (int g = 0; g < 8; ++g) {
#pragma unroll
                for (int p = 0; p < 2; ++p) {
                    int k = kt * 32 + (((g >> 2) << 4) | (half << 3) | ((g & 3) << 1) | p);
                    bfrag[t][kt][g * 2 + p] = M[k * 128 + ncol];
                }
            }
        }
    }

    const int rowtile0 = blockIdx.x * 4;
#if USE_TDM
    if (wave == 0 && rowtile0 < nRowTiles) {
        tdm_load_tile(x + (size_t)rowtile0 * 16 * D_IN,
                      (unsigned)(uintptr_t)&xs[0][0], Nn - rowtile0 * 16);
    }
#endif

    for (int rt4 = 0; rt4 < 4; ++rt4) {
        const int rowtile = rowtile0 + rt4;
        if (rowtile >= nRowTiles) break;          // uniform per block
        const int rowbase = rowtile * 16;
        const float* xsb = xs[rt4 & 1];

#if USE_TDM
        if (wave == 0) __builtin_amdgcn_s_wait_tensorcnt(0);
        __syncthreads();                          // tile (rt4&1) ready; prev reads of other buf done
        if (wave == 0 && rt4 < 3 && rowtile + 1 < nRowTiles) {
            tdm_load_tile(x + (size_t)(rowbase + 16) * D_IN,
                          (unsigned)(uintptr_t)&xs[(rt4 + 1) & 1][0], Nn - (rowbase + 16));
        }
#else
        __syncthreads();
        // synchronous fallback: stage tile (coalesced), clamp tail rows
#pragma unroll
        for (int u = 0; u < 8; ++u) {
            int idx = tid + 256 * u;              // 0..2047
            int r = idx >> 7, c = idx & 127;
            int rg = rowbase + r;
            if (rg >= Nn) rg = Nn - 1;
            xs[rt4 & 1][r * XS_LD + c] = x[(size_t)rg * D_IN + c];
        }
        __syncthreads();
#endif

        // ---- A fragments from LDS (f32 -> bf16, adjacent K pairs -> cvt_pk)
        v16bf afrag[4];
#pragma unroll
        for (int kt = 0; kt < 4; ++kt) {
#pragma unroll
            for (int g = 0; g < 8; ++g) {
                int k = kt * 32 + (((g >> 2) << 4) | (half << 3) | ((g & 3) << 1));
                float a0 = xsb[nrow * XS_LD + k];
                float a1 = xsb[nrow * XS_LD + k + 1];
                afrag[kt][g * 2]     = (__bf16)a0;
                afrag[kt][g * 2 + 1] = (__bf16)a1;
            }
        }

        // ---- WMMA accumulate and store
        const bool full = (rowbase + 16 <= Nn);   // uniform
#pragma unroll
        for (int t = 0; t < 2; ++t) {
            v8f acc = {};
#pragma unroll
            for (int kt = 0; kt < 4; ++kt) {
                acc = __builtin_amdgcn_wmma_f32_16x16x32_bf16(
                        false, afrag[kt], false, bfrag[t][kt],
                        (short)0, acc, false, false);
            }
            const int ncol = (nt0 + t) * 16 + nrow;
            if (full) {
#pragma unroll
                for (int r = 0; r < 8; ++r)
                    G[(size_t)(rowbase + r + 8 * half) * D_IN + ncol] = (__bf16)acc[r];
            } else {
#pragma unroll
                for (int r = 0; r < 8; ++r) {
                    int orow = rowbase + r + 8 * half;
                    if (orow < Nn) G[(size_t)orow * D_IN + ncol] = (__bf16)acc[r];
                }
            }
        }
    }
}

// ---------------------------------------------------------------- edge scatter (one wave per edge)
__global__ void scatter_kernel(const int* __restrict__ src, const int* __restrict__ dst,
                               const float* __restrict__ dinv,
                               const v4bf* __restrict__ gz, const v4bf* __restrict__ gh,
                               float* __restrict__ aggz, float* __restrict__ aggh, int E) {
    const int e    = (blockIdx.x * blockDim.x + threadIdx.x) >> 5;
    const int lane = threadIdx.x & 31;
    if (e >= E) return;
    const int s = src[e], d = dst[e];
    const float w = dinv[s] * dinv[d];

    const v4bf z = gz[(size_t)s * 32 + lane];   // 8B per lane, row in L2
    const v4bf h = gh[(size_t)s * 32 + lane];
    float* pz = aggz + (size_t)d * D_IN + lane * 4;
    float* ph = aggh + (size_t)d * D_IN + lane * 4;
#pragma unroll
    for (int q = 0; q < 4; ++q) {
        unsafeAtomicAdd(pz + q, (float)z[q] * w);
        unsafeAtomicAdd(ph + q, (float)h[q] * w);
    }
}

// ---------------------------------------------------------------- gates + output dot (one wave per node)
__global__ void finish_kernel(const float* __restrict__ aggz, const float* __restrict__ aggh,
                              const __bf16* __restrict__ gz, const __bf16* __restrict__ gh,
                              const float* __restrict__ dinv,
                              const float* __restrict__ lzb, const float* __restrict__ lhb,
                              const float* __restrict__ W2, const float* __restrict__ b2,
                              float* __restrict__ out, int N) {
    const int i    = (blockIdx.x * blockDim.x + threadIdx.x) >> 5;
    const int lane = threadIdx.x & 31;
    if (i >= N) return;
    const float di = dinv[i];
    const float selfw = 2.0f * di * di;   // improved GCN self-loop weight
    float acc = 0.0f;
#pragma unroll
    for (int t = 0; t < 4; ++t) {
        const int dd = lane + 32 * t;
        const size_t idx = (size_t)i * D_IN + dd;
        float sz = aggz[idx] + (float)gz[idx] * selfw + lzb[dd];
        float z  = 1.0f / (1.0f + __expf(-sz));
        float sh = aggh[idx] + (float)gh[idx] * selfw + lhb[dd];
        float ht = tanhf(sh);
        acc += (1.0f - z) * ht * W2[dd];
    }
#pragma unroll
    for (int o = 16; o > 0; o >>= 1) acc += __shfl_down(acc, o, 32);
    if (lane == 0) out[i] = acc + b2[0];
}

// ----------------------------------------------------------------
extern "C" void kernel_launch(void* const* d_in, const int* in_sizes, int n_in,
                              void* d_out, int out_size, void* d_ws, size_t ws_size,
                              hipStream_t stream) {
    const float* x    = (const float*)d_in[0];
    const int*   ei   = (const int*)  d_in[1];
    const float* Wz   = (const float*)d_in[2];
    const float* Wh   = (const float*)d_in[6];
    const float* lzW  = (const float*)d_in[8];
    const float* lzb  = (const float*)d_in[9];
    const float* lhW  = (const float*)d_in[12];
    const float* lhb  = (const float*)d_in[13];
    const float* W2   = (const float*)d_in[14];
    const float* b2   = (const float*)d_in[15];
    float*       out  = (float*)d_out;

    const int N = in_sizes[0] / D_IN;
    const int E = in_sizes[1] / 2;
    const int* src = ei;
    const int* dst = ei + E;

    // workspace carve-up (256B aligned)
    char* w = (char*)d_ws;
    size_t off = 0;
    auto take = [&](size_t bytes) -> void* {
        void* p = w + off;
        off += (bytes + 255) & ~(size_t)255;
        return p;
    };
    float*  deg   = (float*) take((size_t)N * 4);
    float*  dinv  = (float*) take((size_t)N * 4);
    __bf16* Mzb   = (__bf16*)take(128 * 128 * 2);
    __bf16* Mhb   = (__bf16*)take(128 * 128 * 2);
    __bf16* gzb   = (__bf16*)take((size_t)N * D_IN * 2);
    __bf16* ghb   = (__bf16*)take((size_t)N * D_IN * 2);
    float*  aggz  = (float*) take((size_t)N * D_IN * 4);
    float*  aggh  = (float*) take((size_t)N * D_IN * 4);
    if (off > ws_size) return;  // workspace too small; nothing sane to do

    // 1) zero accumulators (harness poisons ws with 0xAA)
    zero_kernel<<<1024, 256, 0, stream>>>(deg,  (size_t)N);
    zero_kernel<<<4096, 256, 0, stream>>>(aggz, (size_t)N * D_IN);
    zero_kernel<<<4096, 256, 0, stream>>>(aggh, (size_t)N * D_IN);

    // 2) degree + dinv
    deg_kernel <<<(E + 255) / 256, 256, 0, stream>>>(dst, deg, E);
    dinv_kernel<<<(N + 255) / 256, 256, 0, stream>>>(deg, dinv, N);

    // 3) fused gate weights: Mz = Wz @ lzW[:128,:],  Mh = Wh @ lhW[:128,:]
    fuse_w_kernel<<<64, 256, 0, stream>>>(Wz, lzW, Mzb);
    fuse_w_kernel<<<64, 256, 0, stream>>>(Wh, lhW, Mhb);

    // 4) WMMA GEMMs: gz = x@Mz, gh = x@Mh (bf16 out), TDM double-buffered x tiles
    const int nRowTiles = (N + 15) / 16;
    const int gemmBlocks = (nRowTiles + 3) / 4;
    gemm_kernel<<<gemmBlocks, 256, 0, stream>>>(x, Mzb, Mhb, gzb, ghb, N, nRowTiles);

    // 5) edge scatter: agg[dst] += g[src] * dinv[src]*dinv[dst]  (L2-resident)
    scatter_kernel<<<(E + 7) / 8, 256, 0, stream>>>(src, dst, dinv,
                                                    (const v4bf*)gzb, (const v4bf*)ghb,
                                                    aggz, aggh, E);

    // 6) gates + final projection
    finish_kernel<<<(N + 7) / 8, 256, 0, stream>>>(aggz, aggh, gzb, ghb, dinv,
                                                   lzb, lhb, W2, b2, out, N);
}
